// PixelAttention_73547019977165
// MI455X (gfx1250) — compile-verified, matrix operand order checked by reading, not run on hardware
//
#include <hip/hip_runtime.h>
#include <hip/hip_bf16.h>

// ---------------- WMMA types & helpers (CDNA5 gfx1250, wave32) ----------------
typedef __attribute__((ext_vector_type(16))) __bf16        v16bf;
typedef __attribute__((ext_vector_type(8)))  float         v8f;
typedef __attribute__((ext_vector_type(4)))  unsigned int  v4u;

union ABReg { v16bf v; v4u q[2]; unsigned int u[8]; };

__device__ __forceinline__ unsigned short f2bf(float f) {
    unsigned int u = __float_as_uint(f);
    u += 0x7FFFu + ((u >> 16) & 1u);           // round-to-nearest-even
    return (unsigned short)(u >> 16);
}
__device__ __forceinline__ float bf2f(unsigned short s) {
    return __uint_as_float(((unsigned int)s) << 16);
}
// ISA 7.12.2 16-bit A 16x32: dword v holds K pair; v=0..3 -> K=8h+{0,2,4,6},
// v=4..7 -> K=16+8h+{0,2,4,6}.  => two contiguous 16B runs per lane.
__device__ __forceinline__ int kmap(int v, int h) {
    return (v < 4 ? 2 * v : 16 + 2 * (v - 4)) + 8 * h;
}

#define KPAD 1952   // 1936 padded to 61*32

// ---------------- packing kernels ----------------
__global__ void zero_fill_kernel(unsigned int* __restrict__ p, int n) {
    int i = blockIdx.x * blockDim.x + threadIdx.x;
    if (i < n) p[i] = 0u;
}

// hc bf16 NHWC [B,44,44,128] from rgb(ch 0..63) + depth(ch 64..127)
__global__ void pack_hc_kernel(const float* __restrict__ rgb, const float* __restrict__ depth,
                               unsigned short* __restrict__ hc, int total) {
    int i = blockIdx.x * blockDim.x + threadIdx.x;
    if (i >= total) return;
    int c = i & 127;
    int pix = (i >> 7) % 1936;
    int b = i / (1936 * 128);
    float v = (c < 64) ? rgb[((size_t)b * 64 + c) * 1936 + pix]
                       : depth[((size_t)b * 64 + (c - 64)) * 1936 + pix];
    hc[i] = f2bf(v);
}

// rgb bf16 NHWC [B,44,44,64]
__global__ void pack_nhwc64_kernel(const float* __restrict__ x, unsigned short* __restrict__ o, int total) {
    int i = blockIdx.x * blockDim.x + threadIdx.x;
    if (i >= total) return;
    int c = i & 63;
    int pix = (i >> 6) % 1936;
    int b = i / (1936 * 64);
    o[i] = f2bf(x[((size_t)b * 64 + c) * 1936 + pix]);
}

// depth -> bf16 rows [b*64+c][KPAD], zero-padded K tail (GEMM B side, K contiguous)
__global__ void pack_depth_kernel(const float* __restrict__ x, unsigned short* __restrict__ o, int total) {
    int i = blockIdx.x * blockDim.x + threadIdx.x;
    if (i >= total) return;
    int p = i % KPAD;
    int row = i / KPAD;
    o[i] = (p < 1936) ? f2bf(x[(size_t)row * 1936 + p]) : (unsigned short)0;
}

// w_s1 [32,1936,3,3] -> rows [m=o*9+t][KPAD] bf16, zero-padded (GEMM A side)
__global__ void pack_ws1_kernel(const float* __restrict__ w, unsigned short* __restrict__ o, int total) {
    int i = blockIdx.x * blockDim.x + threadIdx.x;
    if (i >= total) return;
    int p = i % KPAD;
    int t = (i / KPAD) % 9;
    int oc = i / (KPAD * 9);
    o[i] = (p < 1936) ? f2bf(w[((size_t)oc * 1936 + p) * 9 + t]) : (unsigned short)0;
}

// conv weights -> WMMA B register image: uint dwords [tap][kc][tn][lane][v]
// dword = bf16(k) | bf16(k+1)<<16 with k = kc*32 + kmap(v, lane>>4), n = tn*16 + (lane&15)
// tflip=1: conv-transpose (in/out swap + spatial flip), w layout [in][out][3][3]
__global__ void pack_wB_kernel(const float* __restrict__ w, unsigned int* __restrict__ out,
                               int Cin, int Cout, int tflip, int total) {
    int i = blockIdx.x * blockDim.x + threadIdx.x;
    if (i >= total) return;
    int v = i & 7;
    int lane = (i >> 3) & 31;
    int rest = i >> 8;
    int nTn = Cout >> 4, nKc = Cin >> 5;
    int tn = rest % nTn; rest /= nTn;
    int kc = rest % nKc;
    int tap = rest / nKc;
    int h = lane >> 4;
    int n = tn * 16 + (lane & 15);
    int k = kc * 32 + kmap(v, h);
    int kh = tap / 3, kw = tap % 3;
    float lo, hi;
    if (tflip) {
        lo = w[(((size_t)k * Cout + n) * 3 + (2 - kh)) * 3 + (2 - kw)];
        hi = w[(((size_t)(k + 1) * Cout + n) * 3 + (2 - kh)) * 3 + (2 - kw)];
    } else {
        lo = w[(((size_t)n * Cin + k) * 3 + kh) * 3 + kw];
        hi = w[(((size_t)n * Cin + k + 1) * 3 + kh) * 3 + kw];
    }
    out[i] = (unsigned int)f2bf(lo) | ((unsigned int)f2bf(hi) << 16);
}

// ---------------- W2 GEMM: W2[b, m=o*9+t, c] = sum_p ws1[m,p] * depth[b,c,p] ----------------
// M=288 (18 tiles), N=64 (4 tiles held by one wave), K=KPAD (61 chunks).
// A row loaded once per chunk, reused for 4 WMMAs. Output scattered directly
// into per-batch conv-B register image [b][tap][kc][tn][lane][v].
__global__ void gemm_w2_kernel(const unsigned short* __restrict__ ws1p,
                               const unsigned short* __restrict__ depthb,
                               unsigned int* __restrict__ w2B) {
    int lane = threadIdx.x;
    int m = lane & 15, h = lane >> 4;
    int tm = blockIdx.x, b = blockIdx.z;
    const unsigned short* arow = ws1p + ((size_t)(tm * 16 + m)) * KPAD + 8 * h;
    const unsigned short* brow0 = depthb + ((size_t)b * 64 + m) * KPAD + 8 * h;
    v8f acc[4] = {};
    for (int k0 = 0; k0 < KPAD; k0 += 32) {
        ABReg a;
        a.q[0] = *(const v4u*)(arow + k0);
        a.q[1] = *(const v4u*)(arow + k0 + 16);
#pragma unroll
        for (int j = 0; j < 4; ++j) {
            ABReg bm;
            const unsigned short* brow = brow0 + (size_t)j * 16 * KPAD;
            bm.q[0] = *(const v4u*)(brow + k0);
            bm.q[1] = *(const v4u*)(brow + k0 + 16);
            acc[j] = __builtin_amdgcn_wmma_f32_16x16x32_bf16(false, a.v, false, bm.v,
                                                             (short)0, acc[j], false, false);
        }
    }
    // scatter D into conv-B register image (Cin=64 -> nKc=2, Cout=32 -> nTn=2)
    unsigned short* dst = (unsigned short*)(w2B + (size_t)b * 9 * 2 * 2 * 256);
#pragma unroll
    for (int j = 0; j < 4; ++j) {
#pragma unroll
        for (int r = 0; r < 8; ++r) {
            int mg = tm * 16 + r + 8 * h;
            int o = mg / 9, t = mg % 9;          // conv out-channel, tap
            int cin = j * 16 + (lane & 15);      // conv K
            int kc = cin >> 5, kk = cin & 31;
            int pi = kk >> 1, lohi = kk & 1;
            int h2 = (pi >> 2) & 1;
            int v = (pi & 3) + ((pi & 8) ? 4 : 0);   // inverse of kmap
            int dlane = 16 * h2 + (o & 15);
            int tnc = o >> 4;
            size_t dw = ((((size_t)t * 2 + kc) * 2 + tnc) * 32 + dlane) * 8 + v;
            dst[dw * 2 + lohi] = f2bf(acc[j][r]);
        }
    }
}

// ---------------- generic implicit-GEMM 3x3 conv (bf16 NHWC in, f32 out) ----------------
// wB: WMMA B register image [perBatchW?B:1][9][Cin/32][Cout/16][32][8] dwords
// NT = number of 16-wide out-channel tiles per wave (NT*16 == Cout here):
// A loaded once per K-chunk, reused for NT WMMAs.
template<int NT>
__global__ void conv_wmma_kernel(const unsigned short* __restrict__ in,
                                 const unsigned int* __restrict__ wB,
                                 const float* __restrict__ bias,
                                 float* __restrict__ out,
                                 const unsigned short* __restrict__ zerobuf,
                                 int Hin, int Win, int Cin, int Cout,
                                 int Hout, int Wout, int pad,
                                 int perBatchW, int doRelu, int outNCHW) {
    int lane = threadIdx.x;
    int m = lane & 15, h = lane >> 4;
    int tm = blockIdx.x, b = blockIdx.z;
    int tn0 = blockIdx.y * NT;
    int Pout = Hout * Wout;
    int nKc = Cin >> 5, nTn = Cout >> 4;
    int pm = tm * 16 + m;                 // A row = output pixel
    int py = pm / Wout, px = pm % Wout;
    const unsigned int* wbase = wB + (perBatchW ? (size_t)b * 9 * nKc * nTn * 256 : 0);
    v8f acc[NT] = {};
    for (int tap = 0; tap < 9; ++tap) {
        int kh = tap / 3, kw = tap % 3;
        int sy = py - pad + kh, sx = px - pad + kw;
        bool valid = (pm < Pout) && sy >= 0 && sy < Hin && sx >= 0 && sx < Win;
        const unsigned short* ip = in + (((size_t)b * Hin + sy) * Win + sx) * Cin + 8 * h;
        const unsigned short* abase = valid ? ip : (zerobuf + 8 * h);   // branchless addr select
        const unsigned int* wtap = wbase + ((size_t)tap * nKc) * nTn * 256;
        for (int kc = 0; kc < nKc; ++kc) {
            ABReg a;
            a.q[0] = *(const v4u*)(abase + kc * 32);
            a.q[1] = *(const v4u*)(abase + kc * 32 + 16);
#pragma unroll
            for (int j = 0; j < NT; ++j) {
                ABReg bm;
                const v4u* pw = (const v4u*)(wtap + ((size_t)(kc * nTn + tn0 + j) * 32 + lane) * 8);
                bm.q[0] = pw[0];
                bm.q[1] = pw[1];
                acc[j] = __builtin_amdgcn_wmma_f32_16x16x32_bf16(false, a.v, false, bm.v,
                                                                 (short)0, acc[j], false, false);
            }
        }
    }
#pragma unroll
    for (int j = 0; j < NT; ++j) {
        int o = (tn0 + j) * 16 + m;
        float bv = bias ? bias[o] : 0.f;
#pragma unroll
        for (int r = 0; r < 8; ++r) {
            int p = tm * 16 + r + 8 * h;
            if (p < Pout) {
                float val = acc[j][r] + bv;
                if (doRelu) val = fmaxf(val, 0.f);
                if (outNCHW) {
                    int y = p / Wout, x = p % Wout;
                    out[(((size_t)b * Cout + o) * Hout + y) * Wout + x] = val;
                } else {
                    out[((size_t)b * Pout + p) * Cout + o] = val;
                }
            }
        }
    }
}

// ---------------- BatchNorm (training-mode batch stats) ----------------
__global__ void bn_stats_kernel(const float* __restrict__ y, int n, int Cc,
                                float* __restrict__ mean, float* __restrict__ rstd) {
    __shared__ float ss[256], sq[256];
    int c = blockIdx.x;
    float s = 0.f, q = 0.f;
    for (int i = threadIdx.x; i < n; i += 256) {
        float v = y[(size_t)i * Cc + c];
        s += v; q += v * v;
    }
    ss[threadIdx.x] = s; sq[threadIdx.x] = q;
    __syncthreads();
    for (int st = 128; st > 0; st >>= 1) {
        if ((int)threadIdx.x < st) { ss[threadIdx.x] += ss[threadIdx.x + st]; sq[threadIdx.x] += sq[threadIdx.x + st]; }
        __syncthreads();
    }
    if (threadIdx.x == 0) {
        float mu = ss[0] / n;
        float var = sq[0] / n - mu * mu;     // biased variance (jnp.var)
        mean[c] = mu;
        rstd[c] = rsqrtf(var + 1e-5f);
    }
}

__global__ void bn_relu_kernel(const float* __restrict__ y,
                               const float* __restrict__ mean, const float* __restrict__ rstd,
                               const float* __restrict__ g, const float* __restrict__ be,
                               unsigned short* __restrict__ x, int Cc, int total) {
    int i = blockIdx.x * blockDim.x + threadIdx.x;
    if (i >= total) return;
    int c = i % Cc;
    float v = g[c] * (y[i] - mean[c]) * rstd[c] + be[c];
    x[i] = f2bf(fmaxf(v, 0.f));
}

// ---------------- channel correlation + MLP -> channel attention ----------------
__global__ void corr_mlp_kernel(const float* __restrict__ rgb, const float* __restrict__ depth,
                                const float* __restrict__ w_l1, const float* __restrict__ bl1,
                                const float* __restrict__ w_l2, const float* __restrict__ bl2,
                                float* __restrict__ catt) {
    __shared__ float corr[64];
    __shared__ float hid[16];
    int b = blockIdx.x;
    int c = threadIdx.x;   // 0..63
    const float* r = rgb + ((size_t)b * 64 + c) * 1936;
    const float* d = depth + ((size_t)b * 64 + c) * 1936;
    float s = 0.f;
    for (int p = 0; p < 1936; ++p) s += r[p] * d[p];
    corr[c] = s;
    __syncthreads();
    if (c < 16) {
        float hsum = bl1[c];
        for (int j = 0; j < 64; ++j) hsum += w_l1[c * 64 + j] * corr[j];
        hid[c] = fmaxf(hsum, 0.f);
    }
    __syncthreads();
    float mv = bl2[c];
    for (int j = 0; j < 16; ++j) mv += w_l2[c * 16 + j] * hid[j];
    catt[b * 64 + c] = 1.f / (1.f + __expf(-2.f * mv));   // sigmoid(m + m)
}

// ---------------- conv_t d2 (32 -> 1, 42 -> 44) + sigmoid ----------------
__global__ void convt_d2_kernel(const unsigned short* __restrict__ x3,
                                const float* __restrict__ w_d2, const float* __restrict__ bd2,
                                float* __restrict__ satt) {
    int i = blockIdx.x * blockDim.x + threadIdx.x;
    if (i >= 16 * 1936) return;
    int pix = i % 1936, b = i / 1936;
    int y = pix / 44, x = pix % 44;
    float s = bd2[0];
    for (int kh = 0; kh < 3; ++kh) {
        int sy = y - 2 + kh;
        if (sy < 0 || sy >= 42) continue;
        for (int kw = 0; kw < 3; ++kw) {
            int sx = x - 2 + kw;
            if (sx < 0 || sx >= 42) continue;
            const unsigned short* ip = x3 + (((size_t)b * 42 + sy) * 42 + sx) * 32;
            const float* wv = w_d2 + (2 - kh) * 3 + (2 - kw);   // w_d2[i][0][2-kh][2-kw]
#pragma unroll
            for (int ci = 0; ci < 32; ++ci) s += bf2f(ip[ci]) * wv[ci * 9];
        }
    }
    satt[i] = 1.f / (1.f + __expf(-s));
}

// ---------------- attention multiply + pack h_att bf16 NHWC [B,44,44,128] ----------------
__global__ void mul_att_kernel(const float* __restrict__ hcconv, const float* __restrict__ satt,
                               const float* __restrict__ catt, unsigned short* __restrict__ hatt,
                               int total) {
    int i = blockIdx.x * blockDim.x + threadIdx.x;
    if (i >= total) return;
    int c = i & 127;
    int pix = (i >> 7) % 1936;
    int b = i / (1936 * 128);
    float hv = hcconv[((size_t)b * 1936 + pix) * 64 + (c & 63)];
    float a = (c < 64) ? satt[b * 1936 + pix] : catt[b * 64 + (c - 64)];
    hatt[i] = f2bf(hv * a);
}

// ---------------- launch ----------------
extern "C" void kernel_launch(void* const* d_in, const int* in_sizes, int n_in,
                              void* d_out, int out_size, void* d_ws, size_t ws_size,
                              hipStream_t stream) {
    const float* depth = (const float*)d_in[0];
    const float* rgb   = (const float*)d_in[1];
    const float* w_hc  = (const float*)d_in[2];
    const float* b_hc  = (const float*)d_in[3];
    const float* w_hf  = (const float*)d_in[4];
    const float* b_hf  = (const float*)d_in[5];
    const float* w_s1  = (const float*)d_in[6];
    const float* bs1   = (const float*)d_in[7];
    const float* g1    = (const float*)d_in[8];
    const float* be1   = (const float*)d_in[9];
    const float* w_s2  = (const float*)d_in[10];
    const float* bs2   = (const float*)d_in[11];
    const float* g2    = (const float*)d_in[12];
    const float* be2   = (const float*)d_in[13];
    const float* w_d1  = (const float*)d_in[14];
    const float* bd1   = (const float*)d_in[15];
    const float* g3    = (const float*)d_in[16];
    const float* be3   = (const float*)d_in[17];
    const float* w_d2  = (const float*)d_in[18];
    const float* bd2   = (const float*)d_in[19];
    const float* w_l1  = (const float*)d_in[20];
    const float* bl1   = (const float*)d_in[21];
    const float* w_l2  = (const float*)d_in[22];
    const float* bl2   = (const float*)d_in[23];
    float* outp = (float*)d_out;
    (void)in_sizes; (void)n_in; (void)out_size; (void)ws_size;

    // ---- workspace arena ----
    char* base = (char*)d_ws;
    size_t off = 0;
    auto alloc = [&](size_t bytes) -> void* {
        void* p = base + off;
        off = (off + bytes + 255) & ~(size_t)255;
        return p;
    };
    unsigned short* hc_bf   = (unsigned short*)alloc((size_t)16 * 1936 * 128 * 2);
    unsigned short* rgb_bf  = (unsigned short*)alloc((size_t)16 * 1936 * 64 * 2);
    unsigned short* dep_bf  = (unsigned short*)alloc((size_t)16 * 64 * KPAD * 2);
    unsigned short* ws1p    = (unsigned short*)alloc((size_t)288 * KPAD * 2);
    unsigned int*   w2B     = (unsigned int*)alloc((size_t)16 * 9 * 2 * 2 * 256 * 4);
    unsigned int*   ws2B    = (unsigned int*)alloc((size_t)9 * 1 * 2 * 256 * 4);
    unsigned int*   wd1B    = (unsigned int*)alloc((size_t)9 * 1 * 2 * 256 * 4);
    unsigned int*   whcB    = (unsigned int*)alloc((size_t)9 * 4 * 4 * 256 * 4);
    unsigned int*   whfB    = (unsigned int*)alloc((size_t)9 * 4 * 4 * 256 * 4);
    float*          hcconv  = (float*)alloc((size_t)16 * 1936 * 64 * 4);
    float*          y1      = (float*)alloc((size_t)16 * 1764 * 32 * 4);
    unsigned short* x1      = (unsigned short*)alloc((size_t)16 * 1764 * 32 * 2);
    float*          y2      = (float*)alloc((size_t)16 * 1600 * 32 * 4);
    unsigned short* x2      = (unsigned short*)alloc((size_t)16 * 1600 * 32 * 2);
    float*          y3      = (float*)alloc((size_t)16 * 1764 * 32 * 4);
    unsigned short* x3      = (unsigned short*)alloc((size_t)16 * 1764 * 32 * 2);
    float*          satt    = (float*)alloc((size_t)16 * 1936 * 4);
    float*          catt    = (float*)alloc((size_t)16 * 64 * 4);
    unsigned short* hatt    = (unsigned short*)alloc((size_t)16 * 1936 * 128 * 2);
    float*          mstat   = (float*)alloc(32 * 4 * 6);
    unsigned short* zerobuf = (unsigned short*)alloc(512);     // branchless halo source
    float* mean1 = mstat,       *rstd1 = mstat + 32;
    float* mean2 = mstat + 64,  *rstd2 = mstat + 96;
    float* mean3 = mstat + 128, *rstd3 = mstat + 160;

    const int TB = 256;
    auto gb = [](int n, int t) { return (n + t - 1) / t; };

    // ---- packing ----
    zero_fill_kernel<<<1, 128, 0, stream>>>((unsigned int*)zerobuf, 128);
    {
        int n = 16 * 1936 * 128;
        pack_hc_kernel<<<gb(n, TB), TB, 0, stream>>>(rgb, depth, hc_bf, n);
    }
    {
        int n = 16 * 1936 * 64;
        pack_nhwc64_kernel<<<gb(n, TB), TB, 0, stream>>>(rgb, rgb_bf, n);
    }
    {
        int n = 16 * 64 * KPAD;
        pack_depth_kernel<<<gb(n, TB), TB, 0, stream>>>(depth, dep_bf, n);
    }
    {
        int n = 288 * KPAD;
        pack_ws1_kernel<<<gb(n, TB), TB, 0, stream>>>(w_s1, ws1p, n);
    }
    {
        int n = 9 * 1 * 2 * 256;
        pack_wB_kernel<<<gb(n, TB), TB, 0, stream>>>(w_s2, ws2B, 32, 32, 0, n);
        pack_wB_kernel<<<gb(n, TB), TB, 0, stream>>>(w_d1, wd1B, 32, 32, 1, n);
        n = 9 * 4 * 4 * 256;
        pack_wB_kernel<<<gb(n, TB), TB, 0, stream>>>(w_hc, whcB, 128, 64, 0, n);
        pack_wB_kernel<<<gb(n, TB), TB, 0, stream>>>(w_hf, whfB, 128, 64, 0, n);
    }

    // ---- channel attention (tiny) ----
    corr_mlp_kernel<<<16, 64, 0, stream>>>(rgb, depth, w_l1, bl1, w_l2, bl2, catt);

    // ---- h_c conv: 128->64, pad 1, relu, NHWC fp32 (4 N-tiles per wave) ----
    conv_wmma_kernel<4><<<dim3(121, 1, 16), 32, 0, stream>>>(
        hc_bf, whcB, b_hc, hcconv, zerobuf, 44, 44, 128, 64, 44, 44, 1, 0, 1, 0);

    // ---- W2 = w_s1 x depth (folds pixel_corr + conv_s1 into per-batch weights) ----
    gemm_w2_kernel<<<dim3(18, 1, 16), 32, 0, stream>>>(ws1p, dep_bf, w2B);

    // ---- conv_s1 (per-batch W2): 64->32, valid, 44->42 ----
    conv_wmma_kernel<2><<<dim3(111, 1, 16), 32, 0, stream>>>(
        rgb_bf, w2B, bs1, y1, zerobuf, 44, 44, 64, 32, 42, 42, 0, 1, 0, 0);
    bn_stats_kernel<<<32, 256, 0, stream>>>(y1, 16 * 1764, 32, mean1, rstd1);
    {
        int n = 16 * 1764 * 32;
        bn_relu_kernel<<<gb(n, TB), TB, 0, stream>>>(y1, mean1, rstd1, g1, be1, x1, 32, n);
    }

    // ---- conv_s2: 32->32, valid, 42->40 ----
    conv_wmma_kernel<2><<<dim3(100, 1, 16), 32, 0, stream>>>(
        x1, ws2B, bs2, y2, zerobuf, 42, 42, 32, 32, 40, 40, 0, 0, 0, 0);
    bn_stats_kernel<<<32, 256, 0, stream>>>(y2, 16 * 1600, 32, mean2, rstd2);
    {
        int n = 16 * 1600 * 32;
        bn_relu_kernel<<<gb(n, TB), TB, 0, stream>>>(y2, mean2, rstd2, g2, be2, x2, 32, n);
    }

    // ---- conv_t d1 (as conv with flipped/transposed weights, pad 2): 40->42 ----
    conv_wmma_kernel<2><<<dim3(111, 1, 16), 32, 0, stream>>>(
        x2, wd1B, bd1, y3, zerobuf, 40, 40, 32, 32, 42, 42, 2, 0, 0, 0);
    bn_stats_kernel<<<32, 256, 0, stream>>>(y3, 16 * 1764, 32, mean3, rstd3);
    {
        int n = 16 * 1764 * 32;
        bn_relu_kernel<<<gb(n, TB), TB, 0, stream>>>(y3, mean3, rstd3, g3, be3, x3, 32, n);
    }

    // ---- conv_t d2 + sigmoid -> spatial attention [B,44,44] ----
    {
        int n = 16 * 1936;
        convt_d2_kernel<<<gb(n, TB), TB, 0, stream>>>(x3, w_d2, bd2, satt);
    }

    // ---- attention multiply + pack h_att ----
    {
        int n = 16 * 1936 * 128;
        mul_att_kernel<<<gb(n, TB), TB, 0, stream>>>(hcconv, satt, catt, hatt, n);
    }

    // ---- final conv: 128->64, pad 1, relu, NCHW fp32 -> d_out (4 N-tiles per wave) ----
    conv_wmma_kernel<4><<<dim3(121, 1, 16), 32, 0, stream>>>(
        hatt, whfB, b_hf, outp, zerobuf, 44, 44, 128, 64, 44, 44, 1, 0, 1, 1);
}
